// _MatrixDecomposition2DBase_64759516889199
// MI455X (gfx1250) — matrix operand order repaired and checked by reading, not compiled, not run on hardware
//
#include <hip/hip_runtime.h>
#include <hip/hip_bf16.h>

// Problem constants (from reference): B=16, D=C=512, N=96*96=9216, R=64, 7 steps.
#define B_   16
#define D_   512
#define N_   9216
#define R_   64
#define STEPS_ 7

typedef __attribute__((ext_vector_type(16))) _Float16 v16h;
typedef __attribute__((ext_vector_type(8)))  _Float16 v8h;
typedef __attribute__((ext_vector_type(8)))  float    v8f;

// ---------------------------------------------------------------------------
// WMMA fragment helpers for V_WMMA_F32_16X16X32_F16 (wave32).
// A fragment: row-major [16 x 32] f16, lane m = lane&15, g = lane>>4:
//   halves 0..7  = A[m][k0 + 8g .. k0+8g+7]
//   halves 8..15 = A[m][k0 + 16 + 8g .. k0+16+8g+7]
// B fragment: B is [32 x 16]; memory must hold B transposed ("Bt[col][k]"):
//   lane col = lane&15, halves 0..7 = Bt[col][k0+8g ..], 8..15 = Bt[col][k0+16+8g ..]
// D/C f32: element v <-> row m0 + v + 8g, col = lane&15.
// ---------------------------------------------------------------------------

__device__ __forceinline__ v16h load_a_frag(const _Float16* A, int lda, int k0) {
  const int lane = threadIdx.x & 31;
  const int m = lane & 15;
  const int g = lane >> 4;
  const _Float16* p = A + (size_t)m * lda + (k0 + 8 * g);
  v8h lo = *(const v8h*)(p);
  v8h hi = *(const v8h*)(p + 16);
  return __builtin_shufflevector(lo, hi, 0, 1, 2, 3, 4, 5, 6, 7,
                                 8, 9, 10, 11, 12, 13, 14, 15);
}

__device__ __forceinline__ v16h load_b_frag(const _Float16* Bt, int ldb, int k0) {
  const int lane = threadIdx.x & 31;
  const int n = lane & 15;
  const int g = lane >> 4;
  const _Float16* p = Bt + (size_t)n * ldb + (k0 + 8 * g);
  v8h lo = *(const v8h*)(p);
  v8h hi = *(const v8h*)(p + 16);
  return __builtin_shufflevector(lo, hi, 0, 1, 2, 3, 4, 5, 6, 7,
                                 8, 9, 10, 11, 12, 13, 14, 15);
}

__device__ __forceinline__ v8f wmma32(v16h a, v16h b, v8f c) {
  return __builtin_amdgcn_wmma_f32_16x16x32_f16(
      /*neg_a=*/false, a, /*neg_b=*/false, b,
      /*c_mod=*/(short)0, c, /*reuse_a=*/false, /*reuse_b=*/false);
}

// ---------------------------------------------------------------------------
// x (f32 [B,D,N]) -> x16 (f16 [B,D,N]) and xT16 (f16 [B,N,D]) via LDS transpose.
// ---------------------------------------------------------------------------
__global__ void convert_x_kernel(const float* x, _Float16* x16, _Float16* xT16) {
  __shared__ _Float16 tile[32][33];
  const int b = blockIdx.z;
  const int d0 = blockIdx.y * 32;
  const int n0 = blockIdx.x * 32;
  const int tx = threadIdx.x, ty = threadIdx.y;
  for (int i = ty; i < 32; i += 8) {
    const size_t gi = ((size_t)b * D_ + d0 + i) * N_ + n0 + tx;
    const _Float16 h = (_Float16)x[gi];
    x16[gi] = h;
    tile[i][tx] = h;
  }
  __syncthreads();
  for (int i = ty; i < 32; i += 8) {
    xT16[((size_t)b * N_ + n0 + i) * D_ + d0 + tx] = tile[tx][i];
  }
}

// ---------------------------------------------------------------------------
// bases L2-normalize over D (eps 1e-12), emit f32 + f16 + f16-transposed.
// One block (64 threads = one r each) per batch; coalesced over r.
// ---------------------------------------------------------------------------
__global__ void normalize_bases_kernel(const float* bases_in, float* bases32,
                                       _Float16* bases16, _Float16* basesT16) {
  const int b = blockIdx.x;
  const int r = threadIdx.x;  // 0..63
  const float* src = bases_in + (size_t)b * D_ * R_;
  float s = 0.f;
  for (int d = 0; d < D_; ++d) {
    const float v = src[(size_t)d * R_ + r];
    s += v * v;
  }
  const float inv = 1.0f / fmaxf(sqrtf(s), 1e-12f);
  for (int d = 0; d < D_; ++d) {
    const float v = src[(size_t)d * R_ + r] * inv;
    const size_t idx = ((size_t)b * D_ + d) * R_ + r;
    bases32[idx] = v;
    bases16[idx] = (_Float16)v;
    basesT16[((size_t)b * R_ + r) * D_ + d] = (_Float16)v;
  }
}

// ---------------------------------------------------------------------------
// coef = softmax_r(100 * num[n,:]); emit f32 + f16 + f16-transposed.
// One thread per (b,n); B*N = 147456 threads.
// ---------------------------------------------------------------------------
__global__ void softmax_init_kernel(const float* num, float* coef32,
                                    _Float16* coef16, _Float16* coefT16) {
  const int t = blockIdx.x * 256 + threadIdx.x;  // t = b*N + n
  const float* row = num + (size_t)t * R_;
  float mx = -3.4e38f;
  for (int j = 0; j < R_; ++j) mx = fmaxf(mx, 100.0f * row[j]);
  float s = 0.f;
  for (int j = 0; j < R_; ++j) s += __expf(100.0f * row[j] - mx);
  const float inv = 1.0f / s;
  const int b = t / N_;
  const int n = t - b * N_;
  for (int j = 0; j < R_; ++j) {
    const float c = __expf(100.0f * row[j] - mx) * inv;
    coef32[(size_t)t * R_ + j] = c;
    coef16[(size_t)t * R_ + j] = (_Float16)c;
    coefT16[((size_t)b * R_ + j) * N_ + n] = (_Float16)c;
  }
}

// ---------------------------------------------------------------------------
// GEMM: one wave computes a 16-row x 64-col (= all 4 R tiles) stripe, f32 out.
// C[m, 0..63] = sum_k A[m,k] * Bt[col][k].  grid = (rows/16, B), block = 32.
// Used for num = xT16 @ bases (K=512) and num_b = x16 @ coef (K=9216).
// ---------------------------------------------------------------------------
__global__ void gemm4_kernel(const _Float16* A, long sA, int lda,
                             const _Float16* Bt, long sB, int ldb,
                             float* C, long sC, int K) {
  const int b = blockIdx.y;
  const int m0 = blockIdx.x * 16;
  const _Float16* Ab = A + (size_t)b * sA + (size_t)m0 * lda;
  const _Float16* Bb = Bt + (size_t)b * sB;
  v8f acc[4] = {};
  for (int k0 = 0; k0 < K; k0 += 32) {
    const v16h a = load_a_frag(Ab, lda, k0);
#pragma unroll
    for (int c = 0; c < 4; ++c) {
      const v16h bf = load_b_frag(Bb + (size_t)(c * 16) * ldb, ldb, k0);
      acc[c] = wmma32(a, bf, acc[c]);
    }
  }
  float* Cb = C + (size_t)b * sC + (size_t)m0 * R_;
  const int lane = threadIdx.x & 31;
  const int g = lane >> 4, col = lane & 15;
#pragma unroll
  for (int c = 0; c < 4; ++c)
#pragma unroll
    for (int v = 0; v < 8; ++v)
      Cb[(size_t)(v + 8 * g) * R_ + c * 16 + col] = acc[c][v];
}

// ---------------------------------------------------------------------------
// Single-tile GEMM, f32 output. grid = (Mtiles, ColTiles, B), block = 32.
// Used for reconstruction out = bases16 @ coef^T (K=64).
// ---------------------------------------------------------------------------
__global__ void gemm_tile_f32_kernel(const _Float16* A, long sA, int lda,
                                     const _Float16* Bt, long sB, int ldb,
                                     float* C, long sC, int ldc, int K) {
  const int b = blockIdx.z;
  const int m0 = blockIdx.x * 16;
  const int c0 = blockIdx.y * 16;
  const _Float16* Ab = A + (size_t)b * sA + (size_t)m0 * lda;
  const _Float16* Bb = Bt + (size_t)b * sB + (size_t)c0 * ldb;
  v8f acc = {};
  for (int k0 = 0; k0 < K; k0 += 32)
    acc = wmma32(load_a_frag(Ab, lda, k0), load_b_frag(Bb, ldb, k0), acc);
  const int lane = threadIdx.x & 31;
  const int g = lane >> 4, col = lane & 15;
  float* Cb = C + (size_t)b * sC;
#pragma unroll
  for (int v = 0; v < 8; ++v)
    Cb[(size_t)(m0 + v + 8 * g) * ldc + c0 + col] = acc[v];
}

// ---------------------------------------------------------------------------
// Single-tile GEMM, f16 output. Used for btb = basesT@bases (K=512) and
// ctc = coefT@coef (K=9216); both outputs symmetric 64x64 (values fit f16).
// ---------------------------------------------------------------------------
__global__ void gemm_tile_f16_kernel(const _Float16* A, long sA, int lda,
                                     const _Float16* Bt, long sB, int ldb,
                                     _Float16* C, long sC, int ldc, int K) {
  const int b = blockIdx.z;
  const int m0 = blockIdx.x * 16;
  const int c0 = blockIdx.y * 16;
  const _Float16* Ab = A + (size_t)b * sA + (size_t)m0 * lda;
  const _Float16* Bb = Bt + (size_t)b * sB + (size_t)c0 * ldb;
  v8f acc = {};
  for (int k0 = 0; k0 < K; k0 += 32)
    acc = wmma32(load_a_frag(Ab, lda, k0), load_b_frag(Bb, ldb, k0), acc);
  const int lane = threadIdx.x & 31;
  const int g = lane >> 4, col = lane & 15;
  _Float16* Cb = C + (size_t)b * sC;
#pragma unroll
  for (int v = 0; v < 8; ++v)
    Cb[(size_t)(m0 + v + 8 * g) * ldc + c0 + col] = (_Float16)acc[v];
}

// ---------------------------------------------------------------------------
// Fused multiplicative update: denom = F16 @ S16 (K=64, all 4 col tiles per
// wave), then F = F * numer / (denom + 1e-6); writes f32, f16, f16-transposed.
// Each wave reads only the 16 rows it later overwrites -> no cross-block race.
// grid = (rows/16, B), block = 32.
// ---------------------------------------------------------------------------
__global__ void update_kernel(_Float16* F16, const _Float16* S16,
                              const float* numer, float* F32, _Float16* FT16,
                              int rows) {
  const int b = blockIdx.y;
  const int m0 = blockIdx.x * 16;
  const _Float16* A = F16 + ((size_t)b * rows + m0) * R_;
  const _Float16* Bt = S16 + (size_t)b * R_ * R_;  // symmetric: row==col major
  v8f acc[4] = {};
#pragma unroll
  for (int k0 = 0; k0 < R_; k0 += 32) {
    const v16h a = load_a_frag(A, R_, k0);
#pragma unroll
    for (int c = 0; c < 4; ++c) {
      const v16h bf = load_b_frag(Bt + (size_t)(c * 16) * R_, R_, k0);
      acc[c] = wmma32(a, bf, acc[c]);
    }
  }
  const int lane = threadIdx.x & 31;
  const int g = lane >> 4, col = lane & 15;
#pragma unroll
  for (int c = 0; c < 4; ++c) {
#pragma unroll
    for (int v = 0; v < 8; ++v) {
      const int m = m0 + v + 8 * g;
      const int r = c * 16 + col;
      const size_t idx = ((size_t)b * rows + m) * R_ + r;
      const float dn = acc[c][v];
      const float nv = F32[idx] * numer[idx] / (dn + 1e-6f);
      F32[idx] = nv;
      F16[idx] = (_Float16)nv;
      FT16[((size_t)b * R_ + r) * (size_t)rows + m] = (_Float16)nv;
    }
  }
}

// ---------------------------------------------------------------------------
extern "C" void kernel_launch(void* const* d_in, const int* in_sizes, int n_in,
                              void* d_out, int out_size, void* d_ws, size_t ws_size,
                              hipStream_t stream) {
  (void)in_sizes; (void)n_in; (void)out_size;
  const float* x = (const float*)d_in[0];
  const float* bases_in = (const float*)d_in[1];
  float* out = (float*)d_out;

  char* ws = (char*)d_ws;
  size_t off = 0;
  auto take = [&](size_t bytes) -> char* {
    char* p = ws + off;
    off += (bytes + 255) & ~(size_t)255;
    return p;
  };
  _Float16* x16     = (_Float16*)take((size_t)B_ * D_ * N_ * 2);  // [B,D,N]
  _Float16* xT16    = (_Float16*)take((size_t)B_ * D_ * N_ * 2);  // [B,N,D]
  float*    coef32  = (float*)take((size_t)B_ * N_ * R_ * 4);     // [B,N,R]
  _Float16* coef16  = (_Float16*)take((size_t)B_ * N_ * R_ * 2);  // [B,N,R]
  _Float16* coefT16 = (_Float16*)take((size_t)B_ * N_ * R_ * 2);  // [B,R,N]
  float*    num32   = (float*)take((size_t)B_ * N_ * R_ * 4);     // [B,N,R]
  float*    bases32 = (float*)take((size_t)B_ * D_ * R_ * 4);     // [B,D,R]
  _Float16* bases16 = (_Float16*)take((size_t)B_ * D_ * R_ * 2);  // [B,D,R]
  _Float16* basesT16= (_Float16*)take((size_t)B_ * D_ * R_ * 2);  // [B,R,D]
  float*    numb32  = (float*)take((size_t)B_ * D_ * R_ * 4);     // [B,D,R]
  _Float16* btb16   = (_Float16*)take((size_t)B_ * R_ * R_ * 2);  // [B,R,R]
  _Float16* ctc16   = (_Float16*)take((size_t)B_ * R_ * R_ * 2);  // [B,R,R]
  if (ws_size < off) return;  // defensive: workspace too small

  // One-time: f16 copies of x (both orientations), normalized bases.
  convert_x_kernel<<<dim3(N_ / 32, D_ / 32, B_), dim3(32, 8), 0, stream>>>(x, x16, xT16);
  normalize_bases_kernel<<<B_, R_, 0, stream>>>(bases_in, bases32, bases16, basesT16);

  // coef init: num = x^T @ bases, coef = softmax(100 * num).
  gemm4_kernel<<<dim3(N_ / 16, B_), 32, 0, stream>>>(
      xT16, (long)N_ * D_, D_, basesT16, (long)R_ * D_, D_, num32, (long)N_ * R_, D_);
  softmax_init_kernel<<<(B_ * N_) / 256, 256, 0, stream>>>(num32, coef32, coef16, coefT16);

  // 7 NMF steps + final (differentiable) coef-only update.
  for (int s = 0; s <= STEPS_; ++s) {
    // btb = bases^T @ bases  (f16, symmetric)
    gemm_tile_f16_kernel<<<dim3(R_ / 16, R_ / 16, B_), 32, 0, stream>>>(
        basesT16, (long)R_ * D_, D_, basesT16, (long)R_ * D_, D_,
        btb16, (long)R_ * R_, R_, D_);
    // num = x^T @ bases
    gemm4_kernel<<<dim3(N_ / 16, B_), 32, 0, stream>>>(
        xT16, (long)N_ * D_, D_, basesT16, (long)R_ * D_, D_, num32, (long)N_ * R_, D_);
    // coef = coef * num / (coef @ btb + eps)
    update_kernel<<<dim3(N_ / 16, B_), 32, 0, stream>>>(
        coef16, btb16, num32, coef32, coefT16, N_);
    if (s == STEPS_) break;
    // ctc = coef^T @ coef  (f16, symmetric)
    gemm_tile_f16_kernel<<<dim3(R_ / 16, R_ / 16, B_), 32, 0, stream>>>(
        coefT16, (long)R_ * N_, N_, coefT16, (long)R_ * N_, N_,
        ctc16, (long)R_ * R_, R_, N_);
    // num_b = x @ coef
    gemm4_kernel<<<dim3(D_ / 16, B_), 32, 0, stream>>>(
        x16, (long)D_ * N_, N_, coefT16, (long)R_ * N_, N_, numb32, (long)D_ * R_, N_);
    // bases = bases * num_b / (bases @ ctc + eps)
    update_kernel<<<dim3(D_ / 16, B_), 32, 0, stream>>>(
        bases16, ctc16, numb32, bases32, basesT16, D_);
  }

  // out = bases @ coef^T  -> f32 [B, D, N] == [B, C, H, W]
  gemm_tile_f32_kernel<<<dim3(D_ / 16, N_ / 16, B_), 32, 0, stream>>>(
      bases16, (long)D_ * R_, R_, coef16, (long)N_ * R_, R_,
      out, (long)D_ * N_, N_, R_);
}
